// AttentionBlock_2619930051209
// MI455X (gfx1250) — matrix operand order repaired and checked by reading, not compiled, not run on hardware
//
#include <hip/hip_runtime.h>
#include <math.h>

#define TT     2048
#define HH     2048
#define NHEADS 32
#define NKVH   8
#define DHEAD  64
#define SWIN   128
#define QKVD   3072   // DHEAD * (NHEADS + 2*NKVH)

typedef __bf16 bf16x16 __attribute__((ext_vector_type(16)));
typedef __bf16 bf16x8  __attribute__((ext_vector_type(8)));
typedef float  f32x8   __attribute__((ext_vector_type(8)));
typedef unsigned int u32x4 __attribute__((ext_vector_type(4)));
typedef int  i32x4 __attribute__((ext_vector_type(4)));
typedef int  i32x8 __attribute__((ext_vector_type(8)));

union AFrag { bf16x16 v; bf16x8 h[2]; unsigned u[8]; };

static __device__ inline f32x8 wmma_bf16(bf16x16 a, bf16x16 b, f32x8 c) {
  // D = A(16x32 bf16) * B(32x16 bf16) + C(16x16 f32)
  return __builtin_amdgcn_wmma_f32_16x16x32_bf16(false, a, false, b, (short)0, c,
                                                 false, false);
}

#if __has_builtin(__builtin_amdgcn_tensor_load_to_lds)
#define USE_TDM 1
#else
#define USE_TDM 0
#endif

#define TILE_BYTES (64 * 40 * 2)   // one padded 64x32 bf16 LDS tile

#if USE_TDM
// Issue one TDM descriptor: 64 rows x 32 bf16 tile from global (row stride = K
// elements) into LDS byte offset lds_off, padded +4 DWORDs every 16 DWORDs
// (row stride becomes 40 bf16 elements, matching lds_a[.][64][40]).
static __device__ inline void tdm_load_tile_64x32(unsigned lds_off,
                                                  const __bf16* gptr, int K) {
  unsigned long long ga = (unsigned long long)(uintptr_t)gptr;  // byte address
  u32x4 g0;
  g0.x = 1u;                                    // count=1, user descriptor
  g0.y = lds_off;                               // lds_addr (lds_a is at base 0)
  g0.z = (unsigned)(ga & 0xFFFFFFFFu);          // global_addr[31:0]
  g0.w = (unsigned)((ga >> 32) & 0x01FFFFFFu)   // global_addr[56:32]
         | 0x80000000u;                         // type=2 ("image")
  i32x8 g1;
  g1.s0 = (1 << 16)      // data_size = 1 -> 2 bytes
        | (1 << 20)      // pad_enable
        | (3 << 22)      // pad_interval: 16 DWORDs (one 32-elem bf16 row)
        | (3 << 25);     // pad_amount: 4 DWORDs (8 bf16 elements)
  g1.s1 = (K & 0xFFFF) << 16;                       // tensor_dim0[15:0] @ bits 63:48
  g1.s2 = ((K >> 16) & 0xFFFF) | ((TT & 0xFFFF) << 16); // tensor_dim0 hi, tensor_dim1 lo
  g1.s3 = ((TT >> 16) & 0xFFFF) | (32 << 16);       // tensor_dim1 hi, tile_dim0 = 32
  g1.s4 = 64;                                        // tile_dim1 = 64, tile_dim2 = 0
  g1.s5 = K;                                         // tensor_dim0_stride[31:0]
  g1.s6 = 0;                                         // stride hi, dim1_stride lo
  g1.s7 = 0;
  i32x4 gz = {0, 0, 0, 0};
#if __clang_major__ >= 23
  i32x8 gz8 = {0, 0, 0, 0, 0, 0, 0, 0};
  __builtin_amdgcn_tensor_load_to_lds(g0, g1, gz, gz, gz8, 0);
#else
  __builtin_amdgcn_tensor_load_to_lds(g0, g1, gz, gz, 0);
#endif
}
#endif

// ---------------------------------------------------------------------------
// Kernel 1: RMSNorm (fp32 accumulation), scale, cast to bf16.
// ---------------------------------------------------------------------------
__global__ __launch_bounds__(256)
void rmsnorm_kernel(const __bf16* __restrict__ x, const float* __restrict__ g,
                    __bf16* __restrict__ xn) {
  __shared__ float wsum[8];
  const int t = blockIdx.x;
  const int tid = threadIdx.x;
  const __bf16* xr = x + (size_t)t * HH;
  float v[8];
  float ss = 0.f;
#pragma unroll
  for (int i = 0; i < 8; ++i) {
    float f = (float)xr[tid + 256 * i];
    v[i] = f;
    ss += f * f;
  }
#pragma unroll
  for (int mk = 1; mk < 32; mk <<= 1) ss += __shfl_xor(ss, mk);
  if ((tid & 31) == 0) wsum[tid >> 5] = ss;
  __syncthreads();
  float tot = 0.f;
#pragma unroll
  for (int i = 0; i < 8; ++i) tot += wsum[i];
  const float rms = rsqrtf(tot * (1.f / (float)HH) + 1e-5f);
  __bf16* orow = xn + (size_t)t * HH;
#pragma unroll
  for (int i = 0; i < 8; ++i) {
    int c = tid + 256 * i;
    orow[c] = (__bf16)(v[i] * rms * g[c]);
  }
}

// ---------------------------------------------------------------------------
// Kernel 2/5: C[M,N] = A[M,K] @ W[N,K]^T + bias (+ resid). WMMA bf16.
// Block = 256 threads = 8 waves; block tile 64x64; wave -> 32x16 (two WMMAs).
// A tiles double-buffered in LDS; TDM copy overlapped with WMMA compute.
// ---------------------------------------------------------------------------
__global__ __launch_bounds__(256)
void gemm_bias_kernel(const __bf16* __restrict__ A,
                      const __bf16* __restrict__ W,
                      const __bf16* __restrict__ bias,
                      const __bf16* __restrict__ resid,
                      __bf16* __restrict__ C,
                      int N, int K) {
  __shared__ __align__(32) __bf16 lds_a[2][64][40];  // two 64x32 (+pad) tiles
  const int row0 = blockIdx.x * 64;
  const int col0 = blockIdx.y * 64;
  const int tid = threadIdx.x;
  const int w = tid >> 5, lane = tid & 31;
  const int n_off = (w & 3) * 16;
  const int m_off = (w >> 2) * 32;
  const int r = lane & 15, hh = lane >> 4;

  f32x8 acc0 = {0.f, 0.f, 0.f, 0.f, 0.f, 0.f, 0.f, 0.f};
  f32x8 acc1 = acc0;

  const int ar = tid >> 2;         // 0..63
  const int ac = (tid & 3) * 8;    // 0,8,16,24
  const __bf16* wrow = W + (size_t)(col0 + n_off + r) * K;

#if USE_TDM
  // The TDM writes lds_a "behind the compiler's back" (descriptor carries the
  // LDS address as data). Escape the LDS address into an asm that may write
  // memory so loads of lds_a cannot be folded to undef. Fragment loads below
  // still index lds_a directly so they keep addrspace(3) -> ds_load_b128.
  {
    __bf16* esc = &lds_a[0][0][0];
    asm volatile("" : "+v"(esc) :: "memory");
  }
  if (w == 0) tdm_load_tile_64x32(0u, A + (size_t)row0 * K, K);  // tile 0
#endif

  for (int k0 = 0; k0 < K; k0 += 32) {
#if USE_TDM
    const int cur = (k0 >> 5) & 1;
    if (w == 0) __builtin_amdgcn_s_wait_tensorcnt(0);  // current tile landed
    __syncthreads();                                   // publish to all waves
    asm volatile("" ::: "memory");                     // LDS modified by TDM
#else
    const int cur = 0;
    bf16x8 av = *(const bf16x8*)(A + (size_t)(row0 + ar) * K + k0 + ac);
    *(bf16x8*)&lds_a[0][ar][ac] = av;
    __syncthreads();
#endif

    if (k0 + 32 < K)
      __builtin_prefetch((const void*)(wrow + k0 + 32 + hh * 16), 0, 1);

    AFrag a0, a1;
    a0.h[0] = *(const bf16x8*)&lds_a[cur][m_off + r][hh * 16];
    a0.h[1] = *(const bf16x8*)&lds_a[cur][m_off + r][hh * 16 + 8];
    a1.h[0] = *(const bf16x8*)&lds_a[cur][m_off + 16 + r][hh * 16];
    a1.h[1] = *(const bf16x8*)&lds_a[cur][m_off + 16 + r][hh * 16 + 8];
    // B fragment: 16 contiguous bf16 of weight row (lane = N, idx = K)
    bf16x16 bfrag = *(const bf16x16*)(wrow + k0 + hh * 16);

    acc0 = wmma_bf16(a0.v, bfrag, acc0);
    acc1 = wmma_bf16(a1.v, bfrag, acc1);

#if USE_TDM
    // Prefetch next A tile into the other buffer; nobody reads it this iter,
    // and every wave's DS loads above are consumed before it can land.
    if (w == 0 && k0 + 32 < K)
      tdm_load_tile_64x32((unsigned)((1 ^ cur) * TILE_BYTES),
                          A + (size_t)row0 * K + k0 + 32, K);
#else
    __syncthreads();
#endif
  }

  const int col = col0 + n_off + r;
  const float bv = (float)bias[col];
#pragma unroll
  for (int v = 0; v < 8; ++v) {
    int rowA = row0 + m_off + hh * 8 + v;
    int rowB = rowA + 16;
    float oa = acc0[v] + bv;
    float ob = acc1[v] + bv;
    if (resid) {
      oa += (float)resid[(size_t)rowA * N + col];
      ob += (float)resid[(size_t)rowB * N + col];
    }
    C[(size_t)rowA * N + col] = (__bf16)oa;
    C[(size_t)rowB * N + col] = (__bf16)ob;
  }
}

// ---------------------------------------------------------------------------
// Kernel 3: YaRN RoPE in-place on q (heads 0..31) and k (heads 32..39).
// ---------------------------------------------------------------------------
__global__ __launch_bounds__(256)
void rope_kernel(__bf16* __restrict__ qkv) {
  const int t = blockIdx.x;
  const double two_pi = 6.283185307179586;
  const double ln_theta = log(150000.0);
  const double lowv  = 32.0 * log(4096.0 / (32.0 * two_pi)) / ln_theta;
  const double highv = 32.0 * log(4096.0 / (1.0 * two_pi)) / ln_theta;
  const double conc  = 0.1 * log(32.0) + 1.0;
  for (int p = threadIdx.x; p < (NHEADS + NKVH) * 32; p += 256) {
    int head = p >> 5, j = p & 31;
    double freq = pow(150000.0, (double)(2 * j) / 64.0);
    double interp = 1.0 / (32.0 * freq);
    double extrap = 1.0 / freq;
    double ramp = ((double)j - lowv) / (highv - lowv);
    double m = 1.0 - fmin(fmax(ramp, 0.0), 1.0);
    double invf = interp * (1.0 - m) + extrap * m;
    double ang = (double)t * invf;
    double c, s;
    sincos(ang, &c, &s);
    float cf = (float)(c * conc), sf = (float)(s * conc);
    __bf16* b = qkv + (size_t)t * QKVD + head * DHEAD + j;
    float x1 = (float)b[0], x2 = (float)b[32];
    b[0]  = (__bf16)(x1 * cf - x2 * sf);
    b[32] = (__bf16)(x2 * cf + x1 * sf);
  }
}

// ---------------------------------------------------------------------------
// Kernel 4: sliding-window GQA attention with sink logits.
// Grid (T/16, NHEADS/8), block 256 = 8 waves; each wave: one (q-tile, head).
// QK^T: 2 WMMAs per key tile; online softmax; PV: 4 WMMAs per key tile.
// ---------------------------------------------------------------------------
__global__ __launch_bounds__(256)
void attn_kernel(const __bf16* __restrict__ qkv,
                 const __bf16* __restrict__ sinks,
                 __bf16* __restrict__ out) {
  __shared__ __align__(32) __bf16 lds_p[8][16 * 16];     // per-wave P tile
  __shared__ __align__(32) __bf16 lds_vt[8][64 * 32];    // per-wave V^T [d][k]
  const int qt = blockIdx.x;
  const int w = threadIdx.x >> 5, lane = threadIdx.x & 31;
  const int qh = blockIdx.y * 8 + w;
  const int kvh = qh >> 2;
  const int i0 = qt * 16;
  const int r = lane & 15, hh = lane >> 4;
  const int qcol = qh * DHEAD;
  const int kcol = (NHEADS + kvh) * DHEAD;
  const int vcol = (NHEADS + NKVH + kvh) * DHEAD;

  // Q fragments (16x64 -> two 16x32 A fragments), loaded once
  AFrag qa0, qa1;
  {
    const __bf16* qrow = qkv + (size_t)(i0 + r) * QKVD + qcol;
    qa0.h[0] = *(const bf16x8*)(qrow + hh * 16);
    qa0.h[1] = *(const bf16x8*)(qrow + hh * 16 + 8);
    qa1.h[0] = *(const bf16x8*)(qrow + 32 + hh * 16);
    qa1.h[1] = *(const bf16x8*)(qrow + 32 + hh * 16 + 8);
  }

  const f32x8 zacc = {0.f, 0.f, 0.f, 0.f, 0.f, 0.f, 0.f, 0.f};
  f32x8 o0 = zacc, o1 = zacc, o2 = zacc, o3 = zacc;
  float mrun[8], lrun[8];
#pragma unroll
  for (int v = 0; v < 8; ++v) { mrun[v] = -1e30f; lrun[v] = 0.f; }

  int jt0 = i0 - (SWIN - 1);
  if (jt0 < 0) jt0 = 0;
  jt0 >>= 4;

  for (int jt = jt0; jt <= qt; ++jt) {
    const int j0 = jt * 16;

    // K^T B fragments (lane = key index, elem idx = d): contiguous loads
    const __bf16* krow = qkv + (size_t)(j0 + r) * QKVD + kcol;
    bf16x16 kb0 = *(const bf16x16*)(krow + hh * 16);
    bf16x16 kb1 = *(const bf16x16*)(krow + 32 + hh * 16);

    f32x8 s = wmma_bf16(qa0.v, kb0, zacc);
    s = wmma_bf16(qa1.v, kb1, s);

    // online softmax; C layout: lane half hh -> rows hh*8+v, lane&15 -> col
#pragma unroll
    for (int v = 0; v < 8; ++v) {
      float sv = s[v] * 0.125f;  // 1/sqrt(64)
      int qi = i0 + hh * 8 + v;
      int kj = j0 + r;
      bool ok = (kj <= qi) && (qi - kj < SWIN);
      sv = ok ? sv : -1e30f;
      float tmax = sv;
      tmax = fmaxf(tmax, __shfl_xor(tmax, 1));
      tmax = fmaxf(tmax, __shfl_xor(tmax, 2));
      tmax = fmaxf(tmax, __shfl_xor(tmax, 4));
      tmax = fmaxf(tmax, __shfl_xor(tmax, 8));
      float mnew = fmaxf(mrun[v], tmax);
      float alpha = __expf(mrun[v] - mnew);
      float p = (sv > -5e29f) ? __expf(sv - mnew) : 0.f;
      float rs = p;
      rs += __shfl_xor(rs, 1);
      rs += __shfl_xor(rs, 2);
      rs += __shfl_xor(rs, 4);
      rs += __shfl_xor(rs, 8);
      lrun[v] = lrun[v] * alpha + rs;
      mrun[v] = mnew;
      o0[v] *= alpha; o1[v] *= alpha; o2[v] *= alpha; o3[v] *= alpha;
      lds_p[w][(hh * 8 + v) * 16 + r] = (__bf16)p;
    }

    // V tile (16x64) -> LDS transposed [d][k] for contiguous B fragments
    {
      int vr = lane >> 1;           // key row 0..15
      int vc0 = (lane & 1) * 32;    // d offset 0 or 32
      const __bf16* vp = qkv + (size_t)(j0 + vr) * QKVD + vcol + vc0;
      bf16x16 v0 = *(const bf16x16*)(vp);
      bf16x16 v1 = *(const bf16x16*)(vp + 16);
#pragma unroll
      for (int jj = 0; jj < 16; ++jj)
        lds_vt[w][(vc0 + jj) * 32 + vr] = v0[jj];
#pragma unroll
      for (int jj = 0; jj < 16; ++jj)
        lds_vt[w][(vc0 + 16 + jj) * 32 + vr] = v1[jj];
    }

    // P as A fragment, zero-padded K=16..31 (stale V^T cols are finite * 0)
    AFrag pa;
    pa.h[0] = *(const bf16x8*)&lds_p[w][r * 16];
    pa.h[1] = *(const bf16x8*)&lds_p[w][r * 16 + 8];
    if (hh) pa.u[0] = pa.u[1] = pa.u[2] = pa.u[3] = pa.u[4] = pa.u[5] = pa.u[6] = pa.u[7] = 0;

    bf16x16 vb0 = *(const bf16x16*)&lds_vt[w][(0 + r) * 32 + hh * 16];
    bf16x16 vb1 = *(const bf16x16*)&lds_vt[w][(16 + r) * 32 + hh * 16];
    bf16x16 vb2 = *(const bf16x16*)&lds_vt[w][(32 + r) * 32 + hh * 16];
    bf16x16 vb3 = *(const bf16x16*)&lds_vt[w][(48 + r) * 32 + hh * 16];
    o0 = wmma_bf16(pa.v, vb0, o0);
    o1 = wmma_bf16(pa.v, vb1, o1);
    o2 = wmma_bf16(pa.v, vb2, o2);
    o3 = wmma_bf16(pa.v, vb3, o3);
  }

  // sink logit contributes exp(sink - m) to the denominator only
  const float snk = (float)sinks[qh];
#pragma unroll
  for (int v = 0; v < 8; ++v) {
    float l = lrun[v] + __expf(snk - mrun[v]);
    float inv = 1.f / l;
    o0[v] *= inv; o1[v] *= inv; o2[v] *= inv; o3[v] *= inv;
  }

#pragma unroll
  for (int v = 0; v < 8; ++v) {
    int orow = i0 + hh * 8 + v;
    __bf16* op = out + (size_t)orow * (NHEADS * DHEAD) + qcol;
    op[0 + r]  = (__bf16)o0[v];
    op[16 + r] = (__bf16)o1[v];
    op[32 + r] = (__bf16)o2[v];
    op[48 + r] = (__bf16)o3[v];
  }
}

// ---------------------------------------------------------------------------
extern "C" void kernel_launch(void* const* d_in, const int* in_sizes, int n_in,
                              void* d_out, int out_size, void* d_ws, size_t ws_size,
                              hipStream_t stream) {
  (void)in_sizes; (void)n_in; (void)out_size; (void)ws_size;
  const __bf16* x      = (const __bf16*)d_in[0];
  const float*  gscale = (const float*)d_in[1];
  const __bf16* sinks  = (const __bf16*)d_in[2];
  const __bf16* qkv_w  = (const __bf16*)d_in[3];
  const __bf16* qkv_b  = (const __bf16*)d_in[4];
  const __bf16* out_w  = (const __bf16*)d_in[5];
  const __bf16* out_b  = (const __bf16*)d_in[6];
  __bf16* out = (__bf16*)d_out;

  char* ws = (char*)d_ws;
  __bf16* xn   = (__bf16*)ws;                                        // 8 MB
  __bf16* qkv  = (__bf16*)(ws + (size_t)TT * HH * 2);                // 12 MB
  __bf16* attn = (__bf16*)(ws + (size_t)TT * HH * 2 + (size_t)TT * QKVD * 2); // 8 MB

  rmsnorm_kernel<<<TT, 256, 0, stream>>>(x, gscale, xn);
  gemm_bias_kernel<<<dim3(TT / 64, QKVD / 64), 256, 0, stream>>>(
      xn, qkv_w, qkv_b, nullptr, qkv, QKVD, HH);
  rope_kernel<<<TT, 256, 0, stream>>>(qkv);
  attn_kernel<<<dim3(TT / 16, NHEADS / 8), 256, 0, stream>>>(qkv, sinks, attn);
  gemm_bias_kernel<<<dim3(TT / 64, HH / 64), 256, 0, stream>>>(
      attn, out_w, out_b, x, out, HH, NHEADS * DHEAD);
}
